// NEQUIPLayer_48395691492083
// MI455X (gfx1250) — matrix operand order repaired and checked by reading, not compiled, not run on hardware
//
#include <hip/hip_runtime.h>
#include <hip/hip_bf16.h>
#include <math.h>

// ---------------------------------------------------------------------------
// Types for CDNA5 WMMA (wave32, 16x16x32 bf16 -> f32)
// ---------------------------------------------------------------------------
typedef __attribute__((ext_vector_type(16))) __bf16 v16bf;
typedef __attribute__((ext_vector_type(8)))  __bf16 v8bf;   // 16 bytes = b128 load
typedef __attribute__((ext_vector_type(8)))  float  v8f;

#define F_DIM 64

__device__ __forceinline__ float swishf(float v) {
    // fast swish: v * rcp(1+exp(-v)) -> v_exp_f32 + v_add + v_rcp_f32 + v_mul
    return v * __builtin_amdgcn_rcpf(1.0f + __expf(-v));
}

__device__ __forceinline__ int imin(int a, int b) { return a < b ? a : b; }

// ---------------------------------------------------------------------------
// 1. Device-side Clebsch-Gordan table generation (mirrors reference _real_cg)
// ---------------------------------------------------------------------------
struct cplx { double re, im; };
__device__ __forceinline__ cplx cmul(cplx a, cplx b) {
    cplx r; r.re = a.re * b.re - a.im * b.im; r.im = a.re * b.im + a.im * b.re; return r;
}

__device__ double dfact(int n) {
    double r = 1.0;
    for (int i = 2; i <= n; ++i) r *= (double)i;
    return r;
}

__device__ double cg_coeff(int j1, int m1, int j2, int m2, int j3, int m3) {
    if (m1 + m2 != m3) return 0.0;
    int lo = (j1 > j2) ? (j1 - j2) : (j2 - j1);
    if (j3 < lo || j3 > j1 + j2) return 0.0;
    double pref = sqrt((2.0 * j3 + 1.0) * dfact(j3 + j1 - j2) * dfact(j3 - j1 + j2) *
                       dfact(j1 + j2 - j3) / dfact(j1 + j2 + j3 + 1));
    pref *= sqrt(dfact(j3 + m3) * dfact(j3 - m3) * dfact(j1 - m1) * dfact(j1 + m1) *
                 dfact(j2 - m2) * dfact(j2 + m2));
    double s = 0.0;
    for (int k = 0; k <= j1 + j2 + j3; ++k) {
        int d0 = j1 + j2 - j3 - k;
        int d1 = j1 - m1 - k;
        int d2 = j2 + m2 - k;
        int d3 = j3 - j2 + m1 + k;
        int d4 = j3 - j1 - m2 + k;
        if (d0 < 0 || d1 < 0 || d2 < 0 || d3 < 0 || d4 < 0) continue;
        double term = 1.0 / (dfact(k) * dfact(d0) * dfact(d1) * dfact(d2) * dfact(d3) * dfact(d4));
        s += (k & 1) ? -term : term;
    }
    return pref * s;
}

__device__ void buildU(int l, cplx U[7][7]) {
    for (int i = 0; i < 7; ++i)
        for (int j = 0; j < 7; ++j) { U[i][j].re = 0.0; U[i][j].im = 0.0; }
    double s = 1.0 / sqrt(2.0);
    U[l][l].re = 1.0;
    for (int m = 1; m <= l; ++m) {
        double sgn = (m & 1) ? -1.0 : 1.0;
        U[l + m][l + m].re = sgn * s;
        U[l + m][l - m].re = s;
        U[l - m][l - m].im = s;
        U[l - m][l + m].im = -sgn * s;
    }
}

__device__ void real_cg_dev(int l1, int l2, int l3, float* out) {
    int n1 = 2 * l1 + 1, n2 = 2 * l2 + 1, n3 = 2 * l3 + 1;
    double Cc[105];
    for (int a = 0; a < n1; ++a)
        for (int b = 0; b < n2; ++b)
            for (int c = 0; c < n3; ++c)
                Cc[(a * n2 + b) * n3 + c] =
                    cg_coeff(l1, a - l1, l2, b - l2, l3, c - l3);
    cplx U1[7][7], U2[7][7], U3[7][7];
    buildU(l1, U1); buildU(l2, U2); buildU(l3, U3);
    for (int i = 0; i < n1; ++i)
        for (int j = 0; j < n2; ++j)
            for (int k = 0; k < n3; ++k) {
                double accr = 0.0;
                for (int a = 0; a < n1; ++a)
                    for (int b = 0; b < n2; ++b) {
                        cplx uab = cmul(U1[i][a], U2[j][b]);
                        for (int c = 0; c < n3; ++c) {
                            double cc = Cc[(a * n2 + b) * n3 + c];
                            if (cc == 0.0) continue;
                            cplx u3c; u3c.re = U3[k][c].re; u3c.im = -U3[k][c].im; // conj
                            cplx t = cmul(uab, u3c);
                            accr += t.re * cc;
                        }
                    }
                out[(i * n2 + j) * n3 + k] = (float)accr;
            }
}

// table offsets inside ws (floats)
#define OFF_C110 0     // 3*3*1 = 9
#define OFF_C220 9     // 5*5*1 = 25
#define OFF_C121 34    // 3*5*3 = 45
#define OFF_C211 79    // 5*3*3 = 45
#define OFF_C231 124   // 5*7*3 = 105
#define OFF_C112 229   // 3*3*5 = 45
#define OFF_C132 274   // 3*7*5 = 105
#define OFF_C222 379   // 5*5*5 = 125
#define CG_TOTAL 512   // padded

__global__ void cg_tables_kernel(float* tabs) {
    int t = threadIdx.x;
    switch (t) {
        case 0: real_cg_dev(1, 1, 0, tabs + OFF_C110); break;
        case 1: real_cg_dev(2, 2, 0, tabs + OFF_C220); break;
        case 2: real_cg_dev(1, 2, 1, tabs + OFF_C121); break;
        case 3: real_cg_dev(2, 1, 1, tabs + OFF_C211); break;
        case 4: real_cg_dev(2, 3, 1, tabs + OFF_C231); break;
        case 5: real_cg_dev(1, 1, 2, tabs + OFF_C112); break;
        case 6: real_cg_dev(1, 3, 2, tabs + OFF_C132); break;
        case 7: real_cg_dev(2, 2, 2, tabs + OFF_C222); break;
        default: break;
    }
}

// ---------------------------------------------------------------------------
// 2a. Pack kernels: strided f32 -> contiguous bf16, K padded to mult. of 32.
//     A_packed[m][k] row-major;  B_packed[n][k] (transposed, K-major).
// ---------------------------------------------------------------------------
__global__ void pack_a_kernel(const float* __restrict__ A, __bf16* __restrict__ dst,
                              int M, int K, int Kp, int sAm, int sAk) {
    long idx = (long)blockIdx.x * blockDim.x + threadIdx.x;
    if (idx >= (long)M * Kp) return;
    int m = (int)(idx / Kp);
    int k = (int)(idx % Kp);
    float v = (k < K) ? A[m * sAm + k * sAk] : 0.0f;
    dst[idx] = (__bf16)v;
}

__global__ void pack_b_kernel(const float* __restrict__ B, __bf16* __restrict__ dst,
                              int N, int K, int Kp, int sBk, int sBn) {
    long idx = (long)blockIdx.x * blockDim.x + threadIdx.x;
    if (idx >= (long)N * Kp) return;
    int n = (int)(idx / Kp);
    int k = (int)(idx % Kp);
    float v = (k < K) ? B[k * sBk + n * sBn] : 0.0f;
    dst[n * Kp + k] = (__bf16)v;
}

// ---------------------------------------------------------------------------
// 2b. Packed WMMA GEMM: C = act(scale * A @ B)
//     One wave per 32x64 tile (2x4 WMMA subtiles). Operands pre-packed bf16:
//     A[M][Kp] row-major, Bt[N][Kp] K-major. Per K-step: 12x b128 loads,
//     8x v_wmma_f32_16x16x32_bf16 (1.5 loads/wmma). Kp always a multiple of
//     32 -> guard-free K loop; M/N edges via address clamp + store guard.
//     __launch_bounds__(32, 1): single-wave workgroups; parallelism comes
//     from tens of thousands of blocks, so release the full VGPR budget to
//     let the compiler software-pipeline without spilling accumulators.
// ---------------------------------------------------------------------------
__global__ void __launch_bounds__(32, 1)
wmma_gemm_packed(const __bf16* __restrict__ A,
                 const __bf16* __restrict__ Bt,
                 float* __restrict__ C,
                 int M, int N, int Kp,
                 int sCm, int sCn,
                 float scale, int act) {
    const int tileM = blockIdx.x * 32;
    const int tileN = blockIdx.y * 64;
    const int lane  = threadIdx.x;      // 0..31 (wave32)
    const int half  = lane >> 4;        // 0 or 1
    const int lrow  = lane & 15;

    // clamped per-lane row pointers (A rows / Bt rows = B columns)
    const __bf16* pA[2];
    const __bf16* pB[4];
    #pragma unroll
    for (int t = 0; t < 2; ++t)
        pA[t] = A + (long)imin(tileM + t * 16 + lrow, M - 1) * Kp + half * 8;
    #pragma unroll
    for (int t = 0; t < 4; ++t)
        pB[t] = Bt + (long)imin(tileN + t * 16 + lrow, N - 1) * Kp + half * 8;

    v8f acc[2][4] = {};

    for (int k0 = 0; k0 < Kp; k0 += 32) {
        v16bf a[2], b[4];
        #pragma unroll
        for (int t = 0; t < 2; ++t) {
            // A fragment: element e -> K = k0 + (e>>3)*16 + half*8 + (e&7)
            // = two contiguous 8-element (16 B) chunks at +0 and +16.
            const v8bf* ap = (const v8bf*)(pA[t] + k0);
            v8bf alo = ap[0];
            v8bf ahi = ap[2];   // +16 elements
            a[t] = __builtin_shufflevector(alo, ahi,
                                           0, 1, 2, 3, 4, 5, 6, 7,
                                           8, 9, 10, 11, 12, 13, 14, 15);
        }
        #pragma unroll
        for (int t = 0; t < 4; ++t) {
            // B fragment: element e -> K = k0 + (e>>3)*16 + half*8 + (e&7)
            const v8bf* bp = (const v8bf*)(pB[t] + k0);
            v8bf blo = bp[0];
            v8bf bhi = bp[2];   // +16 elements
            b[t] = __builtin_shufflevector(blo, bhi,
                                           0, 1, 2, 3, 4, 5, 6, 7,
                                           8, 9, 10, 11, 12, 13, 14, 15);
        }
        #pragma unroll
        for (int mi = 0; mi < 2; ++mi)
            #pragma unroll
            for (int ni = 0; ni < 4; ++ni)
                acc[mi][ni] = __builtin_amdgcn_wmma_f32_16x16x32_bf16(
                    false, a[mi], false, b[ni], (short)0, acc[mi][ni], false, false);
    }

    // Store: D VGPR r holds M = sub + r + half*8, N = sub + lrow
    #pragma unroll
    for (int mi = 0; mi < 2; ++mi) {
        #pragma unroll
        for (int r = 0; r < 8; ++r) {
            int m = tileM + mi * 16 + r + half * 8;
            if (m < M) {
                #pragma unroll
                for (int ni = 0; ni < 4; ++ni) {
                    int n = tileN + ni * 16 + lrow;
                    if (n < N) {
                        float v = acc[mi][ni][r] * scale;
                        if (act) v = swishf(v);
                        C[m * sCm + n * sCn] = v;
                    }
                }
            }
        }
    }
}

// ---------------------------------------------------------------------------
// 3. Per-edge geometry: radial basis (8) + real spherical harmonics l=1..3 (15)
// ---------------------------------------------------------------------------
__global__ void edge_geom_kernel(const float* __restrict__ vec,
                                 float* __restrict__ radial,
                                 float* __restrict__ sh, int E) {
    int e = blockIdx.x * blockDim.x + threadIdx.x;
    if (e >= E) return;
    float vx = vec[e * 3 + 0], vy = vec[e * 3 + 1], vz = vec[e * 3 + 2];
    float r2 = vx * vx + vy * vy + vz * vz;
    float x  = sqrtf(fmaxf(r2, 1e-12f));
    float iv = __builtin_amdgcn_rcpf(x);
    float ux = vx * iv, uy = vy * iv, uz = vz * iv;

    float env = 0.0f;
    if (x < 1.0f) {
        float x2 = x * x, x3 = x2 * x;
        float x6 = x3 * x3, x7 = x6 * x, x8 = x7 * x;
        env = 1.0f - 28.0f * x6 + 48.0f * x7 - 21.0f * x8;
    }
    const float sqrt2 = 1.41421356237f;
    #pragma unroll
    for (int n = 1; n <= 8; ++n) {
        radial[e * 8 + (n - 1)] =
            sqrt2 * __sinf(3.14159265358979f * (float)n * x) * iv * env;
    }

    float* sp = sh + e * 15;
    const float s3 = 1.73205080757f;
    sp[0] = s3 * uy; sp[1] = s3 * uz; sp[2] = s3 * ux;
    const float s15 = 3.87298334621f, s5 = 2.2360679775f;
    sp[3] = s15 * ux * uy;
    sp[4] = s15 * uy * uz;
    sp[5] = 0.5f * s5 * (3.0f * uz * uz - 1.0f);
    sp[6] = s15 * ux * uz;
    sp[7] = 0.5f * s15 * (ux * ux - uy * uy);
    const float c35 = 2.09165006634f;   // sqrt(35/8)
    const float c105 = 10.2469507659f;  // sqrt(105)
    const float c21 = 1.62018517461f;   // sqrt(21/8)
    const float c7  = 1.32287565553f;   // sqrt(7)/2
    float uz2 = uz * uz;
    sp[8]  = c35 * uy * (3.0f * ux * ux - uy * uy);
    sp[9]  = c105 * ux * uy * uz;
    sp[10] = c21 * uy * (5.0f * uz2 - 1.0f);
    sp[11] = c7 * uz * (5.0f * uz2 - 3.0f);
    sp[12] = c21 * ux * (5.0f * uz2 - 1.0f);
    sp[13] = 0.5f * c105 * uz * (ux * ux - uy * uy);
    sp[14] = c35 * ux * (ux * ux - 3.0f * uy * uy);
}

// ---------------------------------------------------------------------------
// 4. Zero accumulators
// ---------------------------------------------------------------------------
__global__ void zero_kernel(float* __restrict__ p, long n) {
    long i = (long)blockIdx.x * blockDim.x + threadIdx.x;
    if (i < n) p[i] = 0.0f;
}

// ---------------------------------------------------------------------------
// 5. Message kernel: one thread per (edge, feature).
//    CG tables staged in LDS; gathers sender features; tensor products;
//    scales by radial-MLP mix; scatters to receiver via f32 atomics.
// ---------------------------------------------------------------------------
__global__ void message_kernel(const int* __restrict__ senders,
                               const int* __restrict__ receivers,
                               const float* __restrict__ g0,
                               const float* __restrict__ g1,
                               const float* __restrict__ g2,
                               const float* __restrict__ sh,
                               const float* __restrict__ mix,
                               const float* __restrict__ tabs,
                               float* __restrict__ a0,
                               float* __restrict__ a1,
                               float* __restrict__ a2,
                               int E) {
    __shared__ float T[CG_TOTAL];
    for (int i = threadIdx.x; i < CG_TOTAL; i += blockDim.x) T[i] = tabs[i];
    __syncthreads();

    long idx = (long)blockIdx.x * blockDim.x + threadIdx.x;
    if (idx >= (long)E * F_DIM) return;
    int e = (int)(idx >> 6);
    int f = (int)(idx & 63);

    const int s = senders[e];
    const int r = receivers[e];

    const float* shp  = sh  + e * 15;
    const float sh1v[3] = { shp[0], shp[1], shp[2] };
    const float sh2v[5] = { shp[3], shp[4], shp[5], shp[6], shp[7] };
    const float sh3v[7] = { shp[8], shp[9], shp[10], shp[11], shp[12], shp[13], shp[14] };

    const float s0 = g0[s * 64 + f];
    float s1v[3], s2v[5];
    #pragma unroll
    for (int c = 0; c < 3; ++c) s1v[c] = g1[s * 192 + f * 3 + c];
    #pragma unroll
    for (int c = 0; c < 5; ++c) s2v[c] = g2[s * 320 + f * 5 + c];

    const float* mixp = mix + (long)e * 832;     // 13*F
    const float inv_avg = 0.316227766017f;       // 1/sqrt(10)

    // ---- m0 : [s0 | C110(s1,sh1) | C220(s2,sh2)]  -> a0[r, 3F] ----
    {
        float m = s0 * mixp[f];
        atomicAdd(&a0[r * 192 + f], m * inv_avg);

        float acc = 0.0f;
        #pragma unroll
        for (int i = 0; i < 3; ++i)
            #pragma unroll
            for (int j = 0; j < 3; ++j)
                acc += s1v[i] * sh1v[j] * T[OFF_C110 + i * 3 + j];
        atomicAdd(&a0[r * 192 + 64 + f], acc * mixp[64 + f] * inv_avg);

        acc = 0.0f;
        #pragma unroll
        for (int i = 0; i < 5; ++i)
            #pragma unroll
            for (int j = 0; j < 5; ++j)
                acc += s2v[i] * sh2v[j] * T[OFF_C220 + i * 5 + j];
        atomicAdd(&a0[r * 192 + 128 + f], acc * mixp[128 + f] * inv_avg);
    }

    // ---- m1 : 5 chunks x 3 components -> a1[r, 5F, 3] ----
    {
        float out[5][3];
        #pragma unroll
        for (int k = 0; k < 3; ++k) {
            out[0][k] = s1v[k];
            out[1][k] = s0 * sh1v[k];
            float acc2 = 0.0f, acc3 = 0.0f, acc4 = 0.0f;
            #pragma unroll
            for (int i = 0; i < 3; ++i)
                #pragma unroll
                for (int j = 0; j < 5; ++j)
                    acc2 += s1v[i] * sh2v[j] * T[OFF_C121 + (i * 5 + j) * 3 + k];
            #pragma unroll
            for (int i = 0; i < 5; ++i)
                #pragma unroll
                for (int j = 0; j < 3; ++j)
                    acc3 += s2v[i] * sh1v[j] * T[OFF_C211 + (i * 3 + j) * 3 + k];
            #pragma unroll
            for (int i = 0; i < 5; ++i)
                #pragma unroll
                for (int j = 0; j < 7; ++j)
                    acc4 += s2v[i] * sh3v[j] * T[OFF_C231 + (i * 7 + j) * 3 + k];
            out[2][k] = acc2; out[3][k] = acc3; out[4][k] = acc4;
        }
        #pragma unroll
        for (int q = 0; q < 5; ++q) {
            float mv = mixp[192 + q * 64 + f] * inv_avg;
            #pragma unroll
            for (int k = 0; k < 3; ++k)
                atomicAdd(&a1[r * 960 + (q * 64 + f) * 3 + k], out[q][k] * mv);
        }
    }

    // ---- m2 : 5 chunks x 5 components -> a2[r, 5F, 5] ----
    {
        float out[5][5];
        #pragma unroll
        for (int k = 0; k < 5; ++k) {
            out[0][k] = s2v[k];
            out[1][k] = s0 * sh2v[k];
            float acc2 = 0.0f, acc3 = 0.0f, acc4 = 0.0f;
            #pragma unroll
            for (int i = 0; i < 3; ++i)
                #pragma unroll
                for (int j = 0; j < 3; ++j)
                    acc2 += s1v[i] * sh1v[j] * T[OFF_C112 + (i * 3 + j) * 5 + k];
            #pragma unroll
            for (int i = 0; i < 3; ++i)
                #pragma unroll
                for (int j = 0; j < 7; ++j)
                    acc3 += s1v[i] * sh3v[j] * T[OFF_C132 + (i * 7 + j) * 5 + k];
            #pragma unroll
            for (int i = 0; i < 5; ++i)
                #pragma unroll
                for (int j = 0; j < 5; ++j)
                    acc4 += s2v[i] * sh2v[j] * T[OFF_C222 + (i * 5 + j) * 5 + k];
            out[2][k] = acc2; out[3][k] = acc3; out[4][k] = acc4;
        }
        #pragma unroll
        for (int q = 0; q < 5; ++q) {
            float mv = mixp[512 + q * 64 + f] * inv_avg;
            #pragma unroll
            for (int k = 0; k < 5; ++k)
                atomicAdd(&a2[r * 1600 + (q * 64 + f) * 5 + k], out[q][k] * mv);
        }
    }
}

// ---------------------------------------------------------------------------
// 6. Output kernel: gating, skip connections, final concat layout [N, 9F]
// ---------------------------------------------------------------------------
__global__ void output_kernel(const float* __restrict__ d0,
                              const float* __restrict__ d1,
                              const float* __restrict__ d2,
                              const float* __restrict__ sc0,
                              const float* __restrict__ sc1,
                              const float* __restrict__ sc2,
                              float* __restrict__ out, int Nn) {
    long idx = (long)blockIdx.x * blockDim.x + threadIdx.x;
    if (idx >= (long)Nn * F_DIM) return;
    int n = (int)(idx >> 6);
    int f = (int)(idx & 63);

    float x0 = swishf(d0[n * 192 + f]);
    float gate1 = swishf(d0[n * 192 + 64 + f]);
    float gate2 = swishf(d0[n * 192 + 128 + f]);

    out[n * 576 + f] = x0 + sc0[n * 64 + f];
    #pragma unroll
    for (int c = 0; c < 3; ++c)
        out[n * 576 + 64 + f * 3 + c] =
            d1[n * 192 + f * 3 + c] * gate1 + sc1[n * 192 + f * 3 + c];
    #pragma unroll
    for (int c = 0; c < 5; ++c)
        out[n * 576 + 256 + f * 5 + c] =
            d2[n * 320 + f * 5 + c] * gate2 + sc2[n * 320 + f * 5 + c];
}

// ---------------------------------------------------------------------------
// Host launch helpers
// ---------------------------------------------------------------------------
static inline void pack_a(hipStream_t st, const float* A, __bf16* dst,
                          int M, int K, int Kp, int sAm, int sAk) {
    long tot = (long)M * Kp;
    pack_a_kernel<<<(int)((tot + 255) / 256), 256, 0, st>>>(A, dst, M, K, Kp, sAm, sAk);
}
static inline void pack_b(hipStream_t st, const float* B, __bf16* dst,
                          int N, int K, int Kp, int sBk, int sBn) {
    long tot = (long)N * Kp;
    pack_b_kernel<<<(int)((tot + 255) / 256), 256, 0, st>>>(B, dst, N, K, Kp, sBk, sBn);
}
static inline void gemm(hipStream_t st, const __bf16* A, const __bf16* Bt, float* C,
                        int M, int N, int Kp, int sCm, int sCn, float scale, int act) {
    dim3 grid((M + 31) / 32, (N + 63) / 64);
    wmma_gemm_packed<<<grid, 32, 0, st>>>(A, Bt, C, M, N, Kp, sCm, sCn, scale, act);
}

extern "C" void kernel_launch(void* const* d_in, const int* in_sizes, int n_in,
                              void* d_out, int out_size, void* d_ws, size_t ws_size,
                              hipStream_t stream) {
    const float* vectors    = (const float*)d_in[0];
    const float* node_feats = (const float*)d_in[1];
    // d_in[2] = node_specie (all zeros, single specie -> Wsc[0])
    const int*   senders    = (const int*)d_in[3];
    const int*   receivers  = (const int*)d_in[4];
    const float* Wsc0 = (const float*)d_in[5];
    const float* Wsc1 = (const float*)d_in[6];
    const float* Wsc2 = (const float*)d_in[7];
    const float* Wu0  = (const float*)d_in[8];
    const float* Wu1  = (const float*)d_in[9];
    const float* Wu2  = (const float*)d_in[10];
    const float* Wm1  = (const float*)d_in[11];
    const float* Wm2  = (const float*)d_in[12];
    const float* Wm3  = (const float*)d_in[13];
    const float* Wd0  = (const float*)d_in[14];
    const float* Wd1  = (const float*)d_in[15];
    const float* Wd2  = (const float*)d_in[16];
    float* out = (float*)d_out;

    const int E = in_sizes[0] / 3;
    const int N = in_sizes[1] / (9 * F_DIM);

    // ---------------- workspace layout (all offsets in floats, 32B-aligned)
    float* W = (float*)d_ws;
    size_t off = 0;
    auto alloc_f = [&](size_t n) { float* p = W + off; off += (n + 7) & ~(size_t)7; return p; };
    auto alloc_bf = [&](size_t n) { // n bf16 elements
        float* p = W + off; off += ((n + 1) / 2 + 7) & ~(size_t)7; return (__bf16*)p; };

    float* tabs   = alloc_f(CG_TOTAL);
    float* radial = alloc_f((size_t)E * 8);
    float* sh     = alloc_f((size_t)E * 15);
    float* g0     = alloc_f((size_t)N * 64);
    float* g1     = alloc_f((size_t)N * 192);
    float* g2     = alloc_f((size_t)N * 320);
    float* sc0    = alloc_f((size_t)N * 64);
    float* sc1    = alloc_f((size_t)N * 192);
    float* sc2    = alloc_f((size_t)N * 320);
    float* h1     = alloc_f((size_t)E * 64);
    float* h2     = alloc_f((size_t)E * 64);
    float* mix    = alloc_f((size_t)E * 832);
    float* a0     = alloc_f((size_t)N * 192);
    float* a1     = alloc_f((size_t)N * 960);
    float* a2     = alloc_f((size_t)N * 1600);
    float* d0b    = alloc_f((size_t)N * 192);
    float* d1b    = alloc_f((size_t)N * 192);
    float* d2b    = alloc_f((size_t)N * 320);

    // packed bf16 weights (B matrices, transposed to [N][Kp])
    __bf16* wp_sc0 = alloc_bf(64 * 64);
    __bf16* wp_sc1 = alloc_bf(64 * 64);
    __bf16* wp_sc2 = alloc_bf(64 * 64);
    __bf16* wp_u0  = alloc_bf(64 * 64);
    __bf16* wp_u1  = alloc_bf(64 * 64);
    __bf16* wp_u2  = alloc_bf(64 * 64);
    __bf16* wp_m1  = alloc_bf(64 * 32);    // K=8 padded to 32
    __bf16* wp_m2  = alloc_bf(64 * 64);
    __bf16* wp_m3  = alloc_bf(832 * 64);
    __bf16* wp_d0  = alloc_bf(192 * 192);
    __bf16* wp_d1  = alloc_bf(64 * 320);
    __bf16* wp_d2  = alloc_bf(64 * 320);
    // shared packed-A scratch: max M*Kp = E*64
    __bf16* pa     = alloc_bf((size_t)E * 64);
    (void)ws_size;

    const float inv_f   = 0.125f;            // 1/sqrt(64)
    const float inv_8   = 0.353553390593f;   // 1/sqrt(8)
    const float inv_192 = 0.0721687836487f;  // 1/sqrt(192)
    const float inv_320 = 0.0559016994375f;  // 1/sqrt(320)

    // 1. CG tables + edge geometry
    cg_tables_kernel<<<1, 8, 0, stream>>>(tabs);
    edge_geom_kernel<<<(E + 255) / 256, 256, 0, stream>>>(vectors, radial, sh, E);

    // 2. Pack all weight matrices once
    pack_b(stream, Wsc0, wp_sc0, 64, 64, 64, 64, 1);
    pack_b(stream, Wsc1, wp_sc1, 64, 64, 64, 64, 1);
    pack_b(stream, Wsc2, wp_sc2, 64, 64, 64, 64, 1);
    pack_b(stream, Wu0,  wp_u0,  64, 64, 64, 64, 1);
    pack_b(stream, Wu1,  wp_u1,  64, 64, 64, 64, 1);
    pack_b(stream, Wu2,  wp_u2,  64, 64, 64, 64, 1);
    pack_b(stream, Wm1,  wp_m1,  64, 8, 32, 64, 1);
    pack_b(stream, Wm2,  wp_m2,  64, 64, 64, 64, 1);
    pack_b(stream, Wm3,  wp_m3,  832, 64, 64, 832, 1);
    pack_b(stream, Wd0,  wp_d0,  192, 192, 192, 192, 1);
    pack_b(stream, Wd1,  wp_d1,  64, 320, 320, 64, 1);
    pack_b(stream, Wd2,  wp_d2,  64, 320, 320, 64, 1);

    // 3. Node-side linears (each distinct A slice packed once, used twice)
    pack_a(stream, node_feats, pa, N, 64, 64, 576, 1);          // f0
    gemm(stream, pa, wp_sc0, sc0, N, 64, 64, 64, 1, inv_f, 0);
    gemm(stream, pa, wp_u0,  g0,  N, 64, 64, 64, 1, inv_f, 0);
    for (int c = 0; c < 3; ++c) {   // f1[n,f,c] = node_feats[n, 64 + f*3 + c]
        pack_a(stream, node_feats + 64 + c, pa, N, 64, 64, 576, 3);
        gemm(stream, pa, wp_sc1, sc1 + c, N, 64, 64, 192, 3, inv_f, 0);
        gemm(stream, pa, wp_u1,  g1  + c, N, 64, 64, 192, 3, inv_f, 0);
    }
    for (int c = 0; c < 5; ++c) {   // f2[n,f,c] = node_feats[n, 256 + f*5 + c]
        pack_a(stream, node_feats + 256 + c, pa, N, 64, 64, 576, 5);
        gemm(stream, pa, wp_sc2, sc2 + c, N, 64, 64, 320, 5, inv_f, 0);
        gemm(stream, pa, wp_u2,  g2  + c, N, 64, 64, 320, 5, inv_f, 0);
    }

    // 4. Radial MLP (swish fused into GEMM epilogue)
    pack_a(stream, radial, pa, E, 8, 32, 8, 1);
    gemm(stream, pa, wp_m1, h1, E, 64, 32, 64, 1, inv_8, 1);
    pack_a(stream, h1, pa, E, 64, 64, 64, 1);
    gemm(stream, pa, wp_m2, h2, E, 64, 64, 64, 1, inv_f, 1);
    pack_a(stream, h2, pa, E, 64, 64, 64, 1);
    gemm(stream, pa, wp_m3, mix, E, 832, 64, 832, 1, inv_f, 0);

    // 5. Zero accumulators, then scatter messages
    {
        long ztot = (long)N * (192 + 960 + 1600);
        zero_kernel<<<(int)((ztot + 255) / 256), 256, 0, stream>>>(a0, ztot);
    }
    {
        long tot = (long)E * F_DIM;
        message_kernel<<<(int)((tot + 255) / 256), 256, 0, stream>>>(
            senders, receivers, g0, g1, g2, sh, mix, tabs, a0, a1, a2, E);
    }

    // 6. Output linears
    pack_a(stream, a0, pa, N, 192, 192, 192, 1);
    gemm(stream, pa, wp_d0, d0b, N, 192, 192, 192, 1, inv_192, 0);
    for (int c = 0; c < 3; ++c) {
        pack_a(stream, a1 + c, pa, N, 320, 320, 960, 3);
        gemm(stream, pa, wp_d1, d1b + c, N, 64, 320, 192, 3, inv_320, 0);
    }
    for (int c = 0; c < 5; ++c) {
        pack_a(stream, a2 + c, pa, N, 320, 320, 1600, 5);
        gemm(stream, pa, wp_d2, d2b + c, N, 64, 320, 320, 5, inv_320, 0);
    }

    // 7. Gates + skip connections + concat
    {
        long tot = (long)N * F_DIM;
        output_kernel<<<(int)((tot + 255) / 256), 256, 0, stream>>>(
            d0b, d1b, d2b, sc0, sc1, sc2, out, N);
    }
}
